// MultiHeadAttention_48610439856854
// MI455X (gfx1250) — matrix operand order repaired and checked by reading, not compiled, not run on hardware
//
#include <hip/hip_runtime.h>

typedef __attribute__((ext_vector_type(16))) _Float16 v16h;
typedef __attribute__((ext_vector_type(8)))  float    v8f;
typedef __attribute__((ext_vector_type(4)))  unsigned int v4u;
typedef __attribute__((ext_vector_type(8)))  int v8i;
typedef __attribute__((ext_vector_type(4)))  int v4i;

#define EMBED 1024
#define NHEAD 16
#define HDIM  64
#define BATCH 2
#define SEQ   2048
#define MROWS (BATCH*SEQ)

union FragH { v16h v; uint4 q[2]; };

__device__ __forceinline__ uint2 pack4h(float4 f) {
  union { _Float16 h[4]; uint2 u; } r;
  r.h[0] = (_Float16)f.x; r.h[1] = (_Float16)f.y;
  r.h[2] = (_Float16)f.z; r.h[3] = (_Float16)f.w;
  return r.u;
}

// Issue one TDM 2-D tile load: 64 rows x 64 f16, row stride `stride_elems`,
// into LDS at byte offset `lds_off` with 16B pad after each 128B row
// (pad_interval: 2<<4 = 32 DWORDs, pad_amount: 3+1 = 4 DWORDs -> 144B stride).
__device__ __forceinline__ void tdm_load_tile64(
    unsigned lds_off, const void* gsrc, unsigned stride_elems) {
  unsigned long long ga = (unsigned long long)(uintptr_t)gsrc;
  v4u g0;
  g0[0] = 1u;                                       // count=1, user descriptor
  g0[1] = lds_off;                                  // lds_addr (bytes)
  g0[2] = (unsigned)(ga & 0xFFFFFFFFu);             // global_addr[31:0]
  g0[3] = (unsigned)((ga >> 32) & 0x1FFFFFFu)       // global_addr[56:32]
          | (2u << 30);                             // type=2 ("image")
  v8i g1;
  g1[0] = (int)((1u << 16)                          // data_size = 2 bytes
              | (1u << 20)                          // pad_enable
              | (4u << 22)                          // pad_interval = 32 DWORDs
              | (3u << 25));                        // pad_amount  = 4 DWORDs
  g1[1] = 0;                                        // no atomic barrier; dim0 lo16 = 0
  g1[2] = 0x4000;                                   // tensor_dim0 = 1<<30 (no clip)
  g1[3] = (int)(0x4000u | (64u << 16));             // tensor_dim1 = 1<<30; tile_dim0=64
  g1[4] = 64;                                       // tile_dim1 = 64; tile_dim2 = 0
  g1[5] = (int)stride_elems;                        // tensor_dim0_stride (elems)
  g1[6] = 0;
  g1[7] = 0;
  v4i gz = {0, 0, 0, 0};
#if __clang_major__ >= 23
  v8i gz8 = {0, 0, 0, 0, 0, 0, 0, 0};
  __builtin_amdgcn_tensor_load_to_lds(g0, g1, gz, gz, gz8, 0);
#else
  __builtin_amdgcn_tensor_load_to_lds(g0, g1, gz, gz, 0);
#endif
}

// C = A(fp32, Mx1024) * W(fp32, 1024x1024) + bias
// MODE 0: fp32 row-major out.  MODE 1: f16 out scattered [b][h][s][d].
// MODE 2: f16 out scattered [b][h][d][s] (transposed V).
template<int MODE>
__global__ __launch_bounds__(256) void mha_gemm_f16(
    const float* __restrict__ A, const float* __restrict__ W,
    const float* __restrict__ bias, void* __restrict__ Out) {
  __shared__ _Float16 lA[128][40];   // [m][k], padded row stride
  __shared__ _Float16 lB[128][40];   // [n][k] (W transposed on the fly)

  const int t    = threadIdx.x;
  const int lane = t & 31;
  const int w    = t >> 5;
  const int wm   = w & 3;            // 4 waves along M
  const int wn   = w >> 2;           // 2 waves along N
  const int rowBase = blockIdx.y * 128;
  const int colBase = blockIdx.x * 128;

  v8f zero = {0.f,0.f,0.f,0.f,0.f,0.f,0.f,0.f};
  v8f acc[2][4];
  for (int mi = 0; mi < 2; ++mi)
    for (int ni = 0; ni < 4; ++ni) acc[mi][ni] = zero;

  const int ar = t & 127, aseg = t >> 7;       // A staging: row, 16-col segment
  const int bk = t & 31,  bn0  = (t >> 5)*16;  // W staging: k, 16-n segment

  for (int k0 = 0; k0 < EMBED; k0 += 32) {
    { // stage A tile 128x32 fp32 -> f16
      const float* ap = A + (size_t)(rowBase + ar)*EMBED + k0 + aseg*16;
      float4 f0 = ((const float4*)ap)[0];
      float4 f1 = ((const float4*)ap)[1];
      float4 f2 = ((const float4*)ap)[2];
      float4 f3 = ((const float4*)ap)[3];
      uint2 a = pack4h(f0), b = pack4h(f1), c = pack4h(f2), d = pack4h(f3);
      uint4 p0; p0.x=a.x; p0.y=a.y; p0.z=b.x; p0.w=b.y;
      uint4 p1; p1.x=c.x; p1.y=c.y; p1.z=d.x; p1.w=d.y;
      uint4* dst = (uint4*)&lA[ar][aseg*16];
      dst[0] = p0; dst[1] = p1;
    }
    { // stage W tile 32x128 transposed -> lB[n][k]
      const float* wp = W + (size_t)(k0 + bk)*EMBED + colBase + bn0;
      #pragma unroll
      for (int j = 0; j < 4; ++j) {
        float4 f = ((const float4*)wp)[j];
        lB[bn0 + 4*j + 0][bk] = (_Float16)f.x;
        lB[bn0 + 4*j + 1][bk] = (_Float16)f.y;
        lB[bn0 + 4*j + 2][bk] = (_Float16)f.z;
        lB[bn0 + 4*j + 3][bk] = (_Float16)f.w;
      }
    }
    __syncthreads();

    FragH fa[2];
    const int aoff = (lane & 16) ? 8 : 0;  // A frag: lanes16-31 hold K+8
    #pragma unroll
    for (int mi = 0; mi < 2; ++mi) {
      const int arow = wm*32 + mi*16 + (lane & 15);
      fa[mi].q[0] = *(const uint4*)&lA[arow][aoff];
      fa[mi].q[1] = *(const uint4*)&lA[arow][16 + aoff];
    }
    const int boff = (lane & 16) ? 8 : 0;  // B frag: lanes16-31 hold K+8
    #pragma unroll
    for (int ni = 0; ni < 4; ++ni) {
      FragH fb;
      const int bn = wn*64 + ni*16 + (lane & 15);
      fb.q[0] = *(const uint4*)&lB[bn][boff];
      fb.q[1] = *(const uint4*)&lB[bn][boff + 16];
      #pragma unroll
      for (int mi = 0; mi < 2; ++mi)
        acc[mi][ni] = __builtin_amdgcn_wmma_f32_16x16x32_f16(
            false, fa[mi].v, false, fb.v, (short)0, acc[mi][ni], false, false);
    }
    __syncthreads();
  }

  #pragma unroll
  for (int ni = 0; ni < 4; ++ni) {
    const int col = colBase + wn*64 + ni*16 + (lane & 15);
    const float bv = bias[col];
    #pragma unroll
    for (int mi = 0; mi < 2; ++mi) {
      #pragma unroll
      for (int i = 0; i < 8; ++i) {
        const int row = rowBase + wm*32 + mi*16 + i + ((lane & 16) ? 8 : 0);
        const float val = acc[mi][ni][i] + bv;
        if (MODE == 0) {
          ((float*)Out)[(size_t)row*EMBED + col] = val;
        } else {
          const int h = col >> 6, dd = col & 63;
          const int b = row >> 11, s = row & 2047;
          size_t idx;
          if (MODE == 1) idx = ((size_t)((b*NHEAD + h)*SEQ + s))*HDIM + dd;
          else           idx = ((size_t)((b*NHEAD + h)*HDIM + dd))*SEQ + s;
          ((_Float16*)Out)[idx] = (_Float16)val;
        }
      }
    }
  }
}

// Flash attention: Q,K f16 [b][h][s][64], Vt f16 [b][h][64][s] -> O fp32 [b][s][1024]
__global__ __launch_bounds__(128) void mha_flash_attn(
    const _Float16* __restrict__ Q, const _Float16* __restrict__ K,
    const _Float16* __restrict__ Vt, float* __restrict__ O) {
  __shared__ _Float16 lQ[64][72];
  __shared__ _Float16 lK[64][72];      // [key][d]   (TDM-filled, 144B row stride)
  __shared__ _Float16 lV[64][72];      // [d][key]   (TDM-filled, 144B row stride)
  __shared__ _Float16 lP[4][16][72];   // wave-local P transpose buffer

  const int t    = threadIdx.x;
  const int lane = t & 31;
  const int w    = t >> 5;
  const int qb   = blockIdx.x;
  const int bh   = blockIdx.y;
  const int hh   = bh & (NHEAD - 1);
  const int bb   = bh >> 4;

  const _Float16* Qh = Q  + (size_t)bh * SEQ * HDIM;
  const _Float16* Kh = K  + (size_t)bh * SEQ * HDIM;
  const _Float16* Vh = Vt + (size_t)bh * HDIM * SEQ;

  const unsigned ldsK = (unsigned)(uintptr_t)&lK[0][0];  // low 32 bits = LDS offset
  const unsigned ldsV = (unsigned)(uintptr_t)&lV[0][0];

  const int srow = t >> 1, shalf = t & 1;
  { // stage Q tile (64 x 64)
    const uint4* src = (const uint4*)(Qh + (size_t)(qb*64 + srow)*HDIM + shalf*32);
    uint4* dst = (uint4*)&lQ[srow][shalf*32];
    dst[0]=src[0]; dst[1]=src[1]; dst[2]=src[2]; dst[3]=src[3];
  }
  __syncthreads();

  FragH fq[2];
  {
    const int arow = w*16 + (lane & 15);
    const int ao = (lane & 16) ? 8 : 0;
    #pragma unroll
    for (int c = 0; c < 2; ++c) {
      fq[c].q[0] = *(const uint4*)&lQ[arow][c*32 + ao];
      fq[c].q[1] = *(const uint4*)&lQ[arow][c*32 + 16 + ao];
    }
  }

  float m[8], l[8];
  v8f zero = {0.f,0.f,0.f,0.f,0.f,0.f,0.f,0.f};
  v8f o[4];
  #pragma unroll
  for (int i = 0; i < 8; ++i) { m[i] = -1e30f; l[i] = 0.f; }
  #pragma unroll
  for (int jd = 0; jd < 4; ++jd) o[jd] = zero;

  for (int kb = 0; kb < SEQ/64; ++kb) {
    __syncthreads();   // prior reads of lK/lV done before TDM overwrites
    if (w == 0) {
      // Tensor Data Mover: stream K block [64 keys][64 d] and Vt block
      // [64 d][64 keys] straight into LDS (TENSORcnt-tracked).
      tdm_load_tile64(ldsK, Kh + (size_t)(kb*64)*HDIM, HDIM);
      tdm_load_tile64(ldsV, Vh + (size_t)kb*64, SEQ);
      __builtin_amdgcn_s_wait_tensorcnt(0);
    }
    __syncthreads();   // publish TDM-filled tiles to all waves

    // scores: 16 q-rows x 64 keys, accumulate over d=64 (two K=32 WMMAs)
    v8f sc[4];
    #pragma unroll
    for (int j = 0; j < 4; ++j) {
      sc[j] = zero;
      #pragma unroll
      for (int c = 0; c < 2; ++c) {
        FragH fk;
        const int bn = j*16 + (lane & 15);
        const int bo = c*32 + ((lane & 16) ? 8 : 0);
        fk.q[0] = *(const uint4*)&lK[bn][bo];
        fk.q[1] = *(const uint4*)&lK[bn][bo + 16];
        sc[j] = __builtin_amdgcn_wmma_f32_16x16x32_f16(
            false, fq[c].v, false, fk.v, (short)0, sc[j], false, false);
      }
      #pragma unroll
      for (int i = 0; i < 8; ++i) sc[j][i] *= 0.125f;  // 1/sqrt(64)
    }

    // online softmax (row = vgpr i + 8*half; 16 cols spread across half-wave)
    #pragma unroll
    for (int i = 0; i < 8; ++i) {
      float tm = fmaxf(fmaxf(sc[0][i], sc[1][i]), fmaxf(sc[2][i], sc[3][i]));
      #pragma unroll
      for (int msk = 1; msk <= 8; msk <<= 1)
        tm = fmaxf(tm, __shfl_xor(tm, msk, 32));
      const float mn = fmaxf(m[i], tm);
      const float al = __expf(m[i] - mn);
      m[i] = mn;
      float rs = 0.f;
      #pragma unroll
      for (int j = 0; j < 4; ++j) {
        const float p = __expf(sc[j][i] - mn);
        sc[j][i] = p;
        rs += p;
      }
      #pragma unroll
      for (int msk = 1; msk <= 8; msk <<= 1)
        rs += __shfl_xor(rs, msk, 32);
      l[i] = l[i]*al + rs;
      #pragma unroll
      for (int jd = 0; jd < 4; ++jd) o[jd][i] *= al;
    }

    // transpose P through wave-local LDS into A-fragment layout
    #pragma unroll
    for (int j = 0; j < 4; ++j) {
      const int col = j*16 + (lane & 15);
      #pragma unroll
      for (int i = 0; i < 8; ++i) {
        const int row = i + ((lane & 16) ? 8 : 0);
        lP[w][row][col] = (_Float16)sc[j][i];
      }
    }
    asm volatile("s_wait_dscnt 0" ::: "memory");

    // O += P x V  (two K=32 chunks over the 64 keys)
    #pragma unroll
    for (int c = 0; c < 2; ++c) {
      FragH fp;
      const int prow = lane & 15;
      const int po = c*32 + ((lane & 16) ? 8 : 0);
      fp.q[0] = *(const uint4*)&lP[w][prow][po];
      fp.q[1] = *(const uint4*)&lP[w][prow][po + 16];
      #pragma unroll
      for (int jd = 0; jd < 4; ++jd) {
        FragH fv;
        const int vn = jd*16 + (lane & 15);
        const int vo = c*32 + ((lane & 16) ? 8 : 0);
        fv.q[0] = *(const uint4*)&lV[vn][vo];
        fv.q[1] = *(const uint4*)&lV[vn][vo + 16];
        o[jd] = __builtin_amdgcn_wmma_f32_16x16x32_f16(
            false, fp.v, false, fv.v, (short)0, o[jd], false, false);
      }
    }
  }

  // normalize and store fp32 O[b][s][h*64+d]
  #pragma unroll
  for (int jd = 0; jd < 4; ++jd) {
    const int col = hh*HDIM + jd*16 + (lane & 15);
    #pragma unroll
    for (int i = 0; i < 8; ++i) {
      const int qrow = qb*64 + w*16 + i + ((lane & 16) ? 8 : 0);
      O[((size_t)(bb*SEQ + qrow))*EMBED + col] = o[jd][i] / l[i];
    }
  }
}

extern "C" void kernel_launch(void* const* d_in, const int* in_sizes, int n_in,
                              void* d_out, int out_size, void* d_ws, size_t ws_size,
                              hipStream_t stream) {
  const float* q_in = (const float*)d_in[0];
  const float* k_in = (const float*)d_in[1];
  const float* v_in = (const float*)d_in[2];
  const float* Wq = (const float*)d_in[3];
  const float* bq = (const float*)d_in[4];
  const float* Wk = (const float*)d_in[5];
  const float* bk = (const float*)d_in[6];
  const float* Wv = (const float*)d_in[7];
  const float* bv = (const float*)d_in[8];
  const float* Wo = (const float*)d_in[9];
  const float* bo = (const float*)d_in[10];

  char* ws = (char*)d_ws;
  _Float16* Qh = (_Float16*)(ws);
  _Float16* Kh = (_Float16*)(ws + (size_t)8*1024*1024);
  _Float16* Vt = (_Float16*)(ws + (size_t)16*1024*1024);
  float*    Ob = (float*)   (ws + (size_t)24*1024*1024);

  dim3 ggrid(EMBED/128, MROWS/128);  // (8, 32)
  mha_gemm_f16<1><<<ggrid, 256, 0, stream>>>(q_in, Wq, bq, (void*)Qh);
  mha_gemm_f16<1><<<ggrid, 256, 0, stream>>>(k_in, Wk, bk, (void*)Kh);
  mha_gemm_f16<2><<<ggrid, 256, 0, stream>>>(v_in, Wv, bv, (void*)Vt);
  mha_flash_attn<<<dim3(SEQ/64, BATCH*NHEAD), 128, 0, stream>>>(Qh, Kh, Vt, Ob);
  mha_gemm_f16<0><<<ggrid, 256, 0, stream>>>(Ob, Wo, bo, d_out);
}
